// WindowCrossAttention_20993800143078
// MI455X (gfx1250) — compile-verified, matrix operand order checked by reading, not run on hardware
//
#include <hip/hip_runtime.h>
#include <hip/hip_bf16.h>

typedef __attribute__((ext_vector_type(16))) _Float16 v16h;
typedef __attribute__((ext_vector_type(8)))  _Float16 v8h;
typedef __attribute__((ext_vector_type(8)))  float    v8f;

// ---------------------------------------------------------------------------
// WMMA fragment layouts (CDNA5 wave32, CDNA5 ISA §7.12.2):
// 16-bit A (16x32): lane l holds row M=l&15; element e -> col
//   ((e&8)<<1) + ((l>>4)<<3) + (e&7)
// => elements 0..7 are 8 CONSECUTIVE cols at half8, 8..15 at 16+half8:
//    two aligned 16-byte chunks per lane.
// f32 C/D (16x16): lane l holds col N=l&15, VGPR r holds row M=(l>>4)*8+r.
// ---------------------------------------------------------------------------
__device__ __forceinline__ int patcol(int e, int half8) {
  return ((e & 8) << 1) + half8 + (e & 7);
}

// chunked (2 x b128) f16 fragment load, row-major source
__device__ __forceinline__ v16h frag_rm_f16(const _Float16* base, int ld,
                                            int row0, int col0, int lane) {
  const _Float16* p =
      base + (size_t)(row0 + (lane & 15)) * ld + col0 + ((lane >> 4) << 3);
  v8h c0 = *(const v8h*)p;
  v8h c1 = *(const v8h*)(p + 16);
  v16h f;
#pragma unroll
  for (int e = 0; e < 8; ++e) { f[e] = c0[e]; f[e + 8] = c1[e]; }
  return f;
}

// chunked (4 x b128) f32 fragment load + convert, row-major source
__device__ __forceinline__ v16h frag_rm_f32(const float* base, int ld,
                                            int row0, int col0, int lane) {
  const float* p =
      base + (size_t)(row0 + (lane & 15)) * ld + col0 + ((lane >> 4) << 3);
  float4 a0 = ((const float4*)p)[0];
  float4 a1 = ((const float4*)p)[1];
  float4 b0 = ((const float4*)(p + 16))[0];
  float4 b1 = ((const float4*)(p + 16))[1];
  v16h f;
  f[0] = (_Float16)a0.x;  f[1] = (_Float16)a0.y;
  f[2] = (_Float16)a0.z;  f[3] = (_Float16)a0.w;
  f[4] = (_Float16)a1.x;  f[5] = (_Float16)a1.y;
  f[6] = (_Float16)a1.z;  f[7] = (_Float16)a1.w;
  f[8] = (_Float16)b0.x;  f[9] = (_Float16)b0.y;
  f[10] = (_Float16)b0.z; f[11] = (_Float16)b0.w;
  f[12] = (_Float16)b1.x; f[13] = (_Float16)b1.y;
  f[14] = (_Float16)b1.z; f[15] = (_Float16)b1.w;
  return f;
}

// gather-style B fragment from row-major KxN f32 (prep kernel only)
__device__ __forceinline__ v16h frag_kxn_f32(const float* W, int ldn,
                                             int k0, int n0, int lane) {
  const int n = n0 + (lane & 15);
  const int half8 = (lane >> 4) << 3;
  v16h f;
#pragma unroll
  for (int e = 0; e < 16; ++e)
    f[e] = (_Float16)W[(size_t)(k0 + patcol(e, half8)) * ldn + n];
  return f;
}

// pre-swizzled fragment: 32 contiguous bytes per lane
__device__ __forceinline__ v16h frag_pre(const _Float16* F, int fragIdx,
                                         int lane) {
  return *(const v16h*)(F + ((size_t)fragIdx * 32 + lane) * 16);
}

__device__ __forceinline__ v8f wmma_f16(v16h a, v16h b, v8f c) {
  return __builtin_amdgcn_wmma_f32_16x16x32_f16(false, a, false, b,
                                                (short)0, c, false, false);
}

#define NHEADS 6
#define DHEAD  32
#define NKV    128
#define NQ     512
#define CDIM   192
#define QCDIM  96

// ---------------------------------------------------------------------------
// Kernel P: pre-swizzle all three weight matrices into B-fragment layout.
// 216 fragments total: w_kv 6x24, w_q 3x12, w_proj 6x6. 1 thread = 1 lane.
// ---------------------------------------------------------------------------
__global__ void wca_prep_w_kernel(const float* __restrict__ w_kv,
                                  const float* __restrict__ w_q,
                                  const float* __restrict__ w_proj,
                                  _Float16* __restrict__ Fkv,
                                  _Float16* __restrict__ Fq,
                                  _Float16* __restrict__ Fp) {
  const int t = blockIdx.x * 256 + threadIdx.x;   // 0 .. 6911
  const int lane = t & 31;
  const int frag = t >> 5;                        // 0 .. 215
  const float* W; int ldn, kc, nt; _Float16* dst;
  if (frag < 144)      { W = w_kv;   ldn = 384; kc = frag / 24; nt = frag % 24;
                         dst = Fkv + (size_t)frag * 512; }
  else if (frag < 180) { int f = frag - 144;
                         W = w_q;    ldn = 192; kc = f / 12; nt = f % 12;
                         dst = Fq + (size_t)f * 512; }
  else                 { int f = frag - 180;
                         W = w_proj; ldn = 96;  kc = f / 6;  nt = f % 6;
                         dst = Fp + (size_t)f * 512; }
  v16h v = frag_kxn_f32(W, ldn, kc * 32, nt * 16, lane);
  *(v16h*)(dst + lane * 16) = v;
}

// ---------------------------------------------------------------------------
// Kernel 0: bias gather, directly into C-fragment layout
// biasFrag[head][qt:32][kt:8][lane:32][r:8]  (f32)
// ---------------------------------------------------------------------------
__global__ void wca_bias_kernel(const float* __restrict__ table,
                                const int* __restrict__ rel,
                                float* __restrict__ biasFrag) {
  const int t = blockIdx.x * 256 + threadIdx.x;   // 0 .. 49151
  const int lane = t & 31;
  const int frag = t >> 5;                        // head*256 + qt*8 + kt
  const int head = frag >> 8;
  const int qt = (frag >> 3) & 31;
  const int kt = frag & 7;
  const int kcol = kt * 16 + (lane & 15);
  const int m0 = qt * 16 + ((lane >> 4) << 3);
  float vals[8];
#pragma unroll
  for (int r = 0; r < 8; ++r) {
    const int ri = rel[(m0 + r) * NKV + kcol];
    vals[r] = table[ri * NHEADS + head];
  }
  float4* dst = (float4*)(biasFrag + ((size_t)frag * 32 + lane) * 8);
  dst[0] = make_float4(vals[0], vals[1], vals[2], vals[3]);
  dst[1] = make_float4(vals[4], vals[5], vals[6], vals[7]);
}

// ---------------------------------------------------------------------------
// Kernel 1: kvp = kv @ w_kv + b_kv -> K f16 [b][h][k][d], V f16 TRANSPOSED
// [b][h][d][k].  block = 1 batch, 8 waves; wave w owns M-tile w.
// nt loop split: 0..11 -> K (uniform), 12..23 -> V (uniform, packed store)
// ---------------------------------------------------------------------------
__global__ void __launch_bounds__(256)
wca_proj_kv_kernel(const float* __restrict__ kv, const _Float16* __restrict__ Fkv,
                   const float* __restrict__ b_kv,
                   _Float16* __restrict__ Kf, _Float16* __restrict__ Vt) {
  const int lane = threadIdx.x & 31, wave = threadIdx.x >> 5;
  const int b = blockIdx.x;
  const float* src = kv + (size_t)b * NKV * CDIM;

  v16h a[6];
#pragma unroll
  for (int kc = 0; kc < 6; ++kc)
    a[kc] = frag_rm_f32(src, CDIM, wave * 16, kc * 32, lane);

  const int m0 = wave * 16 + ((lane >> 4) << 3);

  // ---- K half: cols 0..191 ----
  for (int nt = 0; nt < 12; ++nt) {
    const int n = nt * 16 + (lane & 15);
    v16h bfr[6];
#pragma unroll
    for (int kc = 0; kc < 6; ++kc) bfr[kc] = frag_pre(Fkv, kc * 24 + nt, lane);
    v8f c;
    const float bb = b_kv[n];
#pragma unroll
    for (int r = 0; r < 8; ++r) c[r] = bb;
#pragma unroll
    for (int kc = 0; kc < 6; ++kc) c = wmma_f16(a[kc], bfr[kc], c);

    const int head = n >> 5, d = n & 31;
    _Float16* dst = Kf + ((size_t)(b * NHEADS + head) * NKV) * DHEAD + d;
#pragma unroll
    for (int r = 0; r < 8; ++r) dst[(size_t)(m0 + r) * DHEAD] = (_Float16)c[r];
  }

  // ---- V half: cols 192..383, stored transposed [b][h][d][k] ----
  for (int nt = 12; nt < 24; ++nt) {
    const int n = nt * 16 + (lane & 15);
    v16h bfr[6];
#pragma unroll
    for (int kc = 0; kc < 6; ++kc) bfr[kc] = frag_pre(Fkv, kc * 24 + nt, lane);
    v8f c;
    const float bb = b_kv[n];
#pragma unroll
    for (int r = 0; r < 8; ++r) c[r] = bb;
#pragma unroll
    for (int kc = 0; kc < 6; ++kc) c = wmma_f16(a[kc], bfr[kc], c);

    const int head = (n - CDIM) >> 5, d = n & 31;
    v8h hv;
#pragma unroll
    for (int r = 0; r < 8; ++r) hv[r] = (_Float16)c[r];
    *(v8h*)(Vt + ((size_t)(b * NHEADS + head) * DHEAD + d) * NKV + m0) = hv;
  }
}

// ---------------------------------------------------------------------------
// Kernel 2: qp = (q @ w_q + b_q) * scale -> f16 head-major [b][h][q][d]
// ---------------------------------------------------------------------------
__global__ void __launch_bounds__(256)
wca_proj_q_kernel(const float* __restrict__ q, const _Float16* __restrict__ Fq,
                  const float* __restrict__ b_q, _Float16* __restrict__ Qp) {
  const int lane = threadIdx.x & 31, wave = threadIdx.x >> 5;
  const int b = blockIdx.x;
  const float* src = q + (size_t)b * NQ * QCDIM;
  const float scale = 0.17677669529663687f;       // 32^-0.5

  for (int i = 0; i < 4; ++i) {
    const int mt = wave * 4 + i;
    v16h a[3];
#pragma unroll
    for (int kc = 0; kc < 3; ++kc)
      a[kc] = frag_rm_f32(src, QCDIM, mt * 16, kc * 32, lane);
    const int m0 = mt * 16 + ((lane >> 4) << 3);
    for (int nt = 0; nt < 12; ++nt) {
      const int n = nt * 16 + (lane & 15);
      v16h bfr[3];
#pragma unroll
      for (int kc = 0; kc < 3; ++kc) bfr[kc] = frag_pre(Fq, kc * 12 + nt, lane);
      v8f c;
      const float bb = b_q[n];
#pragma unroll
      for (int r = 0; r < 8; ++r) c[r] = bb;
#pragma unroll
      for (int kc = 0; kc < 3; ++kc) c = wmma_f16(a[kc], bfr[kc], c);
      const int head = n >> 5, d = n & 31;
      _Float16* dst = Qp + ((size_t)(b * NHEADS + head) * NQ) * DHEAD + d;
#pragma unroll
      for (int r = 0; r < 8; ++r)
        dst[(size_t)(m0 + r) * DHEAD] = (_Float16)(c[r] * scale);
    }
  }
}

// ---------------------------------------------------------------------------
// Kernel 3: fused attention. block = (batch, head, 128-q chunk), 4 waves.
// K and V fragments are mt-invariant -> loaded once per wave.
// ---------------------------------------------------------------------------
__global__ void __launch_bounds__(128)
wca_attn_kernel(const _Float16* __restrict__ Qp, const _Float16* __restrict__ Kf,
                const _Float16* __restrict__ Vt, const float* __restrict__ biasFrag,
                _Float16* __restrict__ O) {
  __shared__ __align__(16) float    sc[4 * 16 * NKV];   // raw scores
  __shared__ __align__(16) _Float16 ah[4 * 16 * NKV];   // softmaxed probs

  const int lane = threadIdx.x & 31, wave = threadIdx.x >> 5;
  const int bid  = blockIdx.x;
  const int b    = bid / (NHEADS * 4);
  const int rem  = bid % (NHEADS * 4);
  const int head = rem >> 2;
  const int qc   = rem & 3;

  const _Float16* Qh = Qp + ((size_t)(b * NHEADS + head)) * NQ * DHEAD;
  const _Float16* Kh = Kf + ((size_t)(b * NHEADS + head)) * NKV * DHEAD;
  const _Float16* Vh = Vt + ((size_t)(b * NHEADS + head)) * DHEAD * NKV;
  const int qbase = qc * 128 + wave * 32;

  float*    scw = sc + wave * 16 * NKV;
  _Float16* ahw = ah + wave * 16 * NKV;

  __builtin_prefetch(Qh + (size_t)qbase * DHEAD, 0, 1);   // global_prefetch_b8

  // hoisted K (QK^T B operand) and V^T (attn@V B operand) fragments
  v16h kfr[8];
#pragma unroll
  for (int kt = 0; kt < 8; ++kt)
    kfr[kt] = frag_rm_f16(Kh, DHEAD, kt * 16, 0, lane);
  v16h vfr[8];                                    // [dt*4 + kc]
#pragma unroll
  for (int j = 0; j < 8; ++j)
    vfr[j] = frag_rm_f16(Vh, NKV, (j >> 2) * 16, (j & 3) * 32, lane);

  for (int mt = 0; mt < 2; ++mt) {
    const int qt   = qc * 8 + wave * 2 + mt;      // global 16-row q tile
    const int mrow = qbase + mt * 16 + ((lane >> 4) << 3);

    // ---- phase 1: scores = Q K^T + bias (8 WMMAs, K = d = 32) ----
    v16h qf = frag_rm_f16(Qh, DHEAD, qbase + mt * 16, 0, lane);
#pragma unroll
    for (int kt = 0; kt < 8; ++kt) {
      v8f c = *(const v8f*)(biasFrag +
                            ((size_t)((head * 32 + qt) * 8 + kt) * 32 + lane) * 8);
      c = wmma_f16(qf, kfr[kt], c);
      float* dst = scw + (size_t)((lane >> 4) << 3) * NKV + kt * 16 + (lane & 15);
#pragma unroll
      for (int r = 0; r < 8; ++r) dst[(size_t)r * NKV] = c[r];
    }
    __syncthreads();

    // ---- phase 2: softmax, 2 lanes per row ----
    {
      float* row = scw + (size_t)(lane >> 1) * NKV + (lane & 1) * 64;
      float mx = -3.0e38f;
#pragma unroll
      for (int i = 0; i < 16; ++i) {
        float4 v = ((const float4*)row)[i];
        mx = fmaxf(mx, fmaxf(fmaxf(v.x, v.y), fmaxf(v.z, v.w)));
      }
      mx = fmaxf(mx, __shfl_xor(mx, 1, 32));
      float sum = 0.f;
#pragma unroll
      for (int i = 0; i < 16; ++i) {
        float4 v = ((const float4*)row)[i];
        sum += __expf(v.x - mx) + __expf(v.y - mx) +
               __expf(v.z - mx) + __expf(v.w - mx);
      }
      sum += __shfl_xor(sum, 1, 32);
      const float inv = 1.0f / sum;
      unsigned* arow =
          (unsigned*)(ahw + (size_t)(lane >> 1) * NKV + (lane & 1) * 64);
#pragma unroll
      for (int i = 0; i < 32; ++i) {
        float2 v = ((const float2*)row)[i];
        union { _Float16 h[2]; unsigned u; } pk;
        pk.h[0] = (_Float16)(__expf(v.x - mx) * inv);
        pk.h[1] = (_Float16)(__expf(v.y - mx) * inv);
        arow[i] = pk.u;
      }
    }
    __syncthreads();

    // ---- phase 3: out = attn @ V ----
#pragma unroll
    for (int dt = 0; dt < 2; ++dt) {
      v8f c;
#pragma unroll
      for (int r = 0; r < 8; ++r) c[r] = 0.0f;
#pragma unroll
      for (int kc = 0; kc < 4; ++kc) {
        v16h af = frag_rm_f16(ahw, NKV, 0, kc * 32, lane);   // ds b128 x2
        c = wmma_f16(af, vfr[dt * 4 + kc], c);
      }
      const int dcol = head * DHEAD + dt * 16 + (lane & 15);
#pragma unroll
      for (int r = 0; r < 8; ++r)
        O[(size_t)(b * NQ + mrow + r) * CDIM + dcol] = (_Float16)c[r];
    }
    __syncthreads();
  }
}

// ---------------------------------------------------------------------------
// Kernel 4: out = O @ w_proj + b_proj   (f32 out)
// ---------------------------------------------------------------------------
__global__ void __launch_bounds__(256)
wca_proj_out_kernel(const _Float16* __restrict__ O,
                    const _Float16* __restrict__ Fp,
                    const float* __restrict__ b_proj, float* __restrict__ out) {
  const int lane = threadIdx.x & 31, wave = threadIdx.x >> 5;
  const int b = blockIdx.x;
  const _Float16* src = O + (size_t)b * NQ * CDIM;

  for (int i = 0; i < 4; ++i) {
    const int mt = wave * 4 + i;
    v16h a[6];
#pragma unroll
    for (int kc = 0; kc < 6; ++kc)
      a[kc] = frag_rm_f16(src, CDIM, mt * 16, kc * 32, lane);
    const int m0 = mt * 16 + ((lane >> 4) << 3);
#pragma unroll
    for (int nt = 0; nt < 6; ++nt) {
      const int n = nt * 16 + (lane & 15);
      v16h bfr[6];
#pragma unroll
      for (int kc = 0; kc < 6; ++kc) bfr[kc] = frag_pre(Fp, kc * 6 + nt, lane);
      v8f c;
      const float bb = b_proj[n];
#pragma unroll
      for (int r = 0; r < 8; ++r) c[r] = bb;
#pragma unroll
      for (int kc = 0; kc < 6; ++kc) c = wmma_f16(a[kc], bfr[kc], c);
#pragma unroll
      for (int r = 0; r < 8; ++r)
        out[(size_t)(b * NQ + m0 + r) * QCDIM + n] = c[r];
    }
  }
}

// ---------------------------------------------------------------------------
extern "C" void kernel_launch(void* const* d_in, const int* in_sizes, int n_in,
                              void* d_out, int out_size, void* d_ws,
                              size_t ws_size, hipStream_t stream) {
  const float* kv     = (const float*)d_in[0];
  const float* q      = (const float*)d_in[1];
  const float* w_kv   = (const float*)d_in[2];
  const float* b_kv   = (const float*)d_in[3];
  const float* w_q    = (const float*)d_in[4];
  const float* b_q    = (const float*)d_in[5];
  const float* w_proj = (const float*)d_in[6];
  const float* b_proj = (const float*)d_in[7];
  const float* table  = (const float*)d_in[8];
  const int*   rel    = (const int*)d_in[9];

  const int B = 256;
  char* ws = (char*)d_ws;
  // workspace layout (bytes)
  float*    biasF = (float*)ws;                   // 1,572,864
  _Float16* Qp  = (_Float16*)(ws + 1572864);      // 50,331,648
  _Float16* Kf  = (_Float16*)(ws + 51904512);     // 12,582,912
  _Float16* Vt  = (_Float16*)(ws + 64487424);     // 12,582,912 (transposed)
  _Float16* Ob  = (_Float16*)(ws + 77070336);     // 50,331,648
  _Float16* Fkv = (_Float16*)(ws + 127401984);    // 147,456
  _Float16* Fq  = (_Float16*)(ws + 127549440);    // 36,864
  _Float16* Fp  = (_Float16*)(ws + 127586304);    // 36,864
  float* out = (float*)d_out;

  wca_prep_w_kernel<<<dim3(27), dim3(256), 0, stream>>>(w_kv, w_q, w_proj,
                                                        Fkv, Fq, Fp);
  wca_bias_kernel<<<dim3(192), dim3(256), 0, stream>>>(table, rel, biasF);
  wca_proj_kv_kernel<<<dim3(B), dim3(256), 0, stream>>>(kv, Fkv, b_kv, Kf, Vt);
  wca_proj_q_kernel<<<dim3(B), dim3(256), 0, stream>>>(q, Fq, b_q, Qp);
  wca_attn_kernel<<<dim3(B * NHEADS * 4), dim3(128), 0, stream>>>(Qp, Kf, Vt,
                                                                  biasF, Ob);
  wca_proj_out_kernel<<<dim3(B), dim3(256), 0, stream>>>(Ob, Fp, b_proj, out);
}
